// Transformer_42640435314850
// MI455X (gfx1250) — matrix operand order repaired and checked by reading, not compiled, not run on hardware
//
#include <hip/hip_runtime.h>
#include <cstdint>
#include <cstddef>

// ---------------------------------------------------------------------------
// Model constants (match reference)
// ---------------------------------------------------------------------------
#define BB     2
#define TT     2048
#define DIM    512
#define DSTATE 512
#define VOCAB  32000
#define NL     4
#define DFF    2048
#define MM     (BB*TT)          // 4096 rows
#define EPSV   1e-5f

// ---------------------------------------------------------------------------
// Vector types for WMMA (gfx1250, wave32)
// ---------------------------------------------------------------------------
typedef __attribute__((ext_vector_type(16))) __bf16 v16bf;
typedef __attribute__((ext_vector_type(8)))  __bf16 v8bf;
typedef __attribute__((ext_vector_type(8)))  float  v8f;
typedef __attribute__((ext_vector_type(4)))  int    v4i;

#define AS1 __attribute__((address_space(1)))
#define AS3 __attribute__((address_space(3)))

#if defined(__gfx1250__) && __has_builtin(__builtin_amdgcn_global_load_async_to_lds_b128)
#define HAVE_ASYNC 1
#else
#define HAVE_ASYNC 0
#endif

// 16-byte global -> LDS copy. Async (ASYNCcnt-tracked, no VGPR round-trip)
// when the gfx1250 builtin is available, plain VGPR bounce otherwise.
// Builtin signature (from toolchain diagnostic): (v4i AS1*, v4i AS3*, imm, imm).
static __device__ __forceinline__ void copy16_g2l(const __bf16* g, __bf16* l) {
#if HAVE_ASYNC
    __builtin_amdgcn_global_load_async_to_lds_b128(
        (AS1 v4i*)(uintptr_t)g, (AS3 v4i*)(uint32_t)(uintptr_t)l, 0, 0);
#else
    *(uint4*)l = *(const uint4*)g;
#endif
}

static __device__ __forceinline__ void wait_async_all() {
#if defined(__gfx1250__)
#if __has_builtin(__builtin_amdgcn_s_wait_asynccnt)
    __builtin_amdgcn_s_wait_asynccnt(0);
#else
    asm volatile("s_wait_asynccnt 0x0" ::: "memory");
#endif
#endif
}

static __device__ __forceinline__ float silu_f(float x) {
    return x / (1.0f + __expf(-x));
}

// ---------------------------------------------------------------------------
// Embedding gather: x[row] = emb[seq[row]]
// ---------------------------------------------------------------------------
__global__ __launch_bounds__(128)
void embed_kernel(const int* __restrict__ seq, const float* __restrict__ emb,
                  float* __restrict__ x) {
    int row = blockIdx.x;
    int tok = seq[row];
    const float4* src = (const float4*)(emb + (size_t)tok * DIM);
    float4* dst = (float4*)(x + (size_t)row * DIM);
    dst[threadIdx.x] = src[threadIdx.x];   // 128 threads * 4 floats = 512
}

// ---------------------------------------------------------------------------
// RMSNorm over last dim (=512), one block per row, 256 threads
// ---------------------------------------------------------------------------
__global__ __launch_bounds__(256)
void rmsnorm_kernel(const float* __restrict__ x, const float* __restrict__ w,
                    float* __restrict__ out) {
    __shared__ float red[256];
    int row = blockIdx.x, tid = threadIdx.x;
    const float* xr = x + (size_t)row * DIM;
    float v0 = xr[tid], v1 = xr[tid + 256];
    red[tid] = v0 * v0 + v1 * v1;
    __syncthreads();
    #pragma unroll
    for (int s = 128; s > 0; s >>= 1) {
        if (tid < s) red[tid] += red[tid + s];
        __syncthreads();
    }
    float sc = rsqrtf(red[0] * (1.0f / DIM) + EPSV);
    float* o = out + (size_t)row * DIM;
    o[tid]       = v0 * sc * w[tid];
    o[tid + 256] = v1 * sc * w[tid + 256];
}

// ---------------------------------------------------------------------------
// Generic fp32 -> bf16 tensor convert (weights pre-pass), float4 granules
// ---------------------------------------------------------------------------
__global__ __launch_bounds__(256)
void conv_kernel(const float* __restrict__ src, __bf16* __restrict__ dst, int n4) {
    int i = blockIdx.x * 256 + threadIdx.x;
    if (i < n4) {
        float4 f = ((const float4*)src)[i];
        __bf16* p = dst + (size_t)i * 4;
        p[0] = (__bf16)f.x; p[1] = (__bf16)f.y;
        p[2] = (__bf16)f.z; p[3] = (__bf16)f.w;
    }
}

// ---------------------------------------------------------------------------
// Tiled WMMA GEMM:  C[M,N] = epilogue( A[M,K](f32) @ W[N,K](bf16)^T )
//   MODE 0: C = acc + bias
//   MODE 1: C = acc + bias + extra          (residual; bias may be null)
//   MODE 2: C = silu(acc)
//   MODE 3: C = extra * acc                 (SwiGLU gate product)
// 256 thr / 8 waves; tile 128(M) x 64(N); K-step 32.
// Double-buffered LDS; B panel staged with async global->LDS b128 copies,
// A panel software-pipelined through VGPRs with fp32->bf16 convert.
// ---------------------------------------------------------------------------
#define BM  128
#define BN  64
#define KT  32
#define LDA 48
#define LDB 48

template <int MODE>
__global__ __launch_bounds__(256)
void gemm_kernel(const float* __restrict__ A, const __bf16* __restrict__ W, int ldw,
                 const float* __restrict__ bias, const float* __restrict__ extra,
                 float* __restrict__ C, int M, int N, int K) {
    __shared__ __bf16 As[2][BM * LDA];
    __shared__ __bf16 Bs[2][BN * LDB];

    const int tid  = threadIdx.x;
    const int lane = tid & 31;
    const int wave = tid >> 5;
    const int wm = wave >> 1;          // 0..3  (M direction)
    const int wn = wave & 1;           // 0..1  (N direction)
    const int bm = blockIdx.y * BM;
    const int bn = blockIdx.x * BN;

    const int r  = lane & 15;
    const int kh = lane >> 4;

    // A staging geometry: 1024 float4 per 128x32 slab, 4 per thread
    const int aj0  = tid;              // + it*256
    // B staging geometry: 64 rows x 32 bf16 = 256 x 16B, 1 per thread
    const int brow = tid >> 2;
    const int bseg = (tid & 3) * 8;    // bf16 elements

    float4 fa[4];

    auto loadA = [&](int kk) {
        #pragma unroll
        for (int it = 0; it < 4; ++it) {
            int j = aj0 + it * 256, row = j >> 3, cg = (j & 7) * 4;
            fa[it] = *(const float4*)(A + (size_t)(bm + row) * K + kk + cg);
        }
    };
    auto storeA = [&](int buf) {
        #pragma unroll
        for (int it = 0; it < 4; ++it) {
            int j = aj0 + it * 256, row = j >> 3, cg = (j & 7) * 4;
            __bf16* p = &As[buf][row * LDA + cg];
            p[0] = (__bf16)fa[it].x; p[1] = (__bf16)fa[it].y;
            p[2] = (__bf16)fa[it].z; p[3] = (__bf16)fa[it].w;
        }
    };
    auto issueB = [&](int buf, int kk) {
        const __bf16* g = W + (size_t)(bn + brow) * ldw + kk + bseg;
        copy16_g2l(g, &Bs[buf][brow * LDB + bseg]);
    };

    v8f acc[2][2] = {};

    // prologue: fill buffer 0
    loadA(0);
    issueB(0, 0);
    storeA(0);
    wait_async_all();
    __syncthreads();

    int cur = 0;
    for (int kk = 0; kk < K; kk += KT) {
        const int nxt = kk + KT;
        const bool has_next = nxt < K;
        if (has_next) {
            loadA(nxt);            // global loads in flight during compute
            issueB(cur ^ 1, nxt);  // async copies in flight during compute
        }

        // ---- A operand: 16x32 bf16 (row=lane%16, K = kh*16 .. +15 contiguous)
        v16bf a[2], b[2];
        #pragma unroll
        for (int i = 0; i < 2; ++i) {
            const __bf16* pa = &As[cur][(wm * 32 + i * 16 + r) * LDA + kh * 16];
            v8bf lo = *(const v8bf*)pa;            // K = kh*16 .. +7
            v8bf hi = *(const v8bf*)(pa + 8);      // K = kh*16+8 .. +15
            #pragma unroll
            for (int e = 0; e < 8; ++e) { a[i][e] = lo[e]; a[i][e + 8] = hi[e]; }
        }
        // ---- B operand: 32x16 bf16 (col=lane%16, K = kh*16 .. +15 contiguous)
        #pragma unroll
        for (int j = 0; j < 2; ++j) {
            const __bf16* pb = &Bs[cur][(wn * 32 + j * 16 + r) * LDB + kh * 16];
            b[j] = *(const v16bf*)pb;
        }
        #pragma unroll
        for (int i = 0; i < 2; ++i)
            #pragma unroll
            for (int j = 0; j < 2; ++j)
                acc[i][j] = __builtin_amdgcn_wmma_f32_16x16x32_bf16(
                    false, a[i], false, b[j], (short)0, acc[i][j], false, false);

        if (has_next) storeA(cur ^ 1);   // waits its loads, converts, ds-stores
        wait_async_all();
        __syncthreads();
        cur ^= 1;
    }

    // ---- epilogue: C/D layout: VGPR rr, lane l -> M = rr + 8*(l>>4), N = l&15
    #pragma unroll
    for (int i = 0; i < 2; ++i) {
        #pragma unroll
        for (int j = 0; j < 2; ++j) {
            int nn    = bn + wn * 32 + j * 16 + r;
            int mbase = bm + wm * 32 + i * 16 + 8 * kh;
            float bv = (MODE <= 1 && bias) ? bias[nn] : 0.0f;
            #pragma unroll
            for (int rr = 0; rr < 8; ++rr) {
                size_t idx = (size_t)(mbase + rr) * N + nn;
                float v = acc[i][j][rr];
                if (MODE == 0)      v = v + bv;
                else if (MODE == 1) v = v + bv + extra[idx];
                else if (MODE == 2) v = silu_f(v);
                else                v = extra[idx] * v;
                C[idx] = v;
            }
        }
    }
}

// ---------------------------------------------------------------------------
// Recurrent scan (both batches packed into WMMA M-rows 0..1; rows 2..15 = 0).
// latent_{t} = rmsnorm( latent_{t-1} @ WABl^T + u2_t ) * wln
// One persistent workgroup: 1024 thr = 32 waves, wave w owns cols [16w,16w+16).
// WABl = bf16 view of WAB[:, :512] (row stride ldw), L2-resident.
// ---------------------------------------------------------------------------
#define LATP 528

__global__ __launch_bounds__(1024)
void scan_kernel(const float* __restrict__ u2, const __bf16* __restrict__ wabl,
                 int ldw, const float* __restrict__ wln, float* __restrict__ lat) {
    __shared__ __bf16 latBf[16 * LATP];
    __shared__ float  sumsq[2];

    const int tid  = threadIdx.x;
    const int lane = tid & 31;
    const int w    = tid >> 5;         // 0..31
    const int r    = lane & 15;
    const int kh   = lane >> 4;
    const int col  = w * 16 + r;       // 0..511

    for (int i = tid; i < 16 * LATP; i += 1024) latBf[i] = (__bf16)0.0f;
    if (tid < 2) sumsq[tid] = 0.0f;
    __syncthreads();

    const float wlnc = wln[col];

    for (int t = 0; t < TT; ++t) {
        // ---- acc = latent(16x512) @ WABl^T  (cols owned by this wave)
        v8f acc = {};
        #pragma unroll 4
        for (int kk = 0; kk < DSTATE; kk += 32) {
            const __bf16* pa = &latBf[r * LATP + kk + kh * 16];
            v8bf lo = *(const v8bf*)pa;
            v8bf hi = *(const v8bf*)(pa + 8);
            v16bf a;
            #pragma unroll
            for (int e = 0; e < 8; ++e) { a[e] = lo[e]; a[e + 8] = hi[e]; }
            v16bf b = *(const v16bf*)(wabl + (size_t)col * ldw + kk + kh * 16);
            acc = __builtin_amdgcn_wmma_f32_16x16x32_bf16(
                false, a, false, b, (short)0, acc, false, false);
        }
        // rows 0/1 of the tile = batches 0/1 (lanes 0..15, VGPRs 0/1)
        float v0 = acc[0] + u2[(size_t)(0 * TT + t) * DSTATE + col];
        float v1 = acc[1] + u2[(size_t)(1 * TT + t) * DSTATE + col];
        float s0 = (kh == 0) ? v0 * v0 : 0.0f;
        float s1 = (kh == 0) ? v1 * v1 : 0.0f;
        #pragma unroll
        for (int m = 16; m >= 1; m >>= 1) {
            s0 += __shfl_xor(s0, m, 32);
            s1 += __shfl_xor(s1, m, 32);
        }
        __syncthreads();                       // all latBf reads done, sumsq zeroed
        if (lane == 0) { atomicAdd(&sumsq[0], s0); atomicAdd(&sumsq[1], s1); }
        __syncthreads();                       // sums complete
        float sc0 = rsqrtf(sumsq[0] * (1.0f / DSTATE) + EPSV);
        float sc1 = rsqrtf(sumsq[1] * (1.0f / DSTATE) + EPSV);
        if (kh == 0) {
            float n0 = v0 * sc0 * wlnc;
            float n1 = v1 * sc1 * wlnc;
            latBf[0 * LATP + col] = (__bf16)n0;
            latBf[1 * LATP + col] = (__bf16)n1;
            lat[(size_t)(0 * TT + t) * DSTATE + col] = n0;
            lat[(size_t)(1 * TT + t) * DSTATE + col] = n1;
        }
        __syncthreads();                       // latBf writes + sumsq reads done
        if (tid < 2) sumsq[tid] = 0.0f;        // re-zero for next step
    }
}

// ---------------------------------------------------------------------------
// Per-row cross-entropy:  rowloss = logsumexp(logits_row) - logits_row[target]
// ---------------------------------------------------------------------------
__global__ __launch_bounds__(256)
void rowloss_kernel(const float* __restrict__ logits, const int* __restrict__ tgt,
                    float* __restrict__ rowloss) {
    __shared__ float red[256];
    int row = blockIdx.x, tid = threadIdx.x;
    const float* lr = logits + (size_t)row * VOCAB;
    float mx = -1e30f;
    for (int i = tid; i < VOCAB; i += 256) mx = fmaxf(mx, lr[i]);
    red[tid] = mx; __syncthreads();
    #pragma unroll
    for (int s = 128; s > 0; s >>= 1) {
        if (tid < s) red[tid] = fmaxf(red[tid], red[tid + s]);
        __syncthreads();
    }
    float Mx = red[0];
    __syncthreads();
    float sm = 0.0f;
    for (int i = tid; i < VOCAB; i += 256) sm += __expf(lr[i] - Mx);
    red[tid] = sm; __syncthreads();
    #pragma unroll
    for (int s = 128; s > 0; s >>= 1) {
        if (tid < s) red[tid] += red[tid + s];
        __syncthreads();
    }
    if (tid == 0) rowloss[row] = (Mx + __logf(red[0])) - lr[tgt[row]];
}

__global__ __launch_bounds__(256)
void loss_reduce_kernel(const float* __restrict__ rowloss, float* __restrict__ out) {
    __shared__ float red[256];
    int tid = threadIdx.x;
    float s = 0.0f;
    for (int i = tid; i < MM; i += 256) s += rowloss[i];
    red[tid] = s; __syncthreads();
    #pragma unroll
    for (int st = 128; st > 0; st >>= 1) {
        if (tid < st) red[tid] += red[tid + st];
        __syncthreads();
    }
    if (tid == 0) out[0] = red[0] * (1.0f / MM);
}

// ---------------------------------------------------------------------------
// Host-side orchestration
// ---------------------------------------------------------------------------
extern "C" void kernel_launch(void* const* d_in, const int* in_sizes, int n_in,
                              void* d_out, int out_size, void* d_ws, size_t ws_size,
                              hipStream_t stream) {
    (void)in_sizes; (void)n_in; (void)out_size; (void)ws_size;

    const int*   seq     = (const int*)d_in[0];
    const int*   targets = (const int*)d_in[1];
    const float* emb     = (const float*)d_in[2];
    const float* ln_f    = (const float*)d_in[3];

    // ---- fp32 workspace carve-up
    float* ws  = (float*)d_ws;
    const size_t SZD = (size_t)MM * DIM;     // 4096*512
    float* X   = ws;
    float* XN  = X   + SZD;
    float* U   = XN  + SZD;
    float* U2  = U   + SZD;
    float* LAT = U2  + SZD;
    float* T3  = LAT + SZD;
    float* H   = T3  + SZD;                  // 4096*2048
    float* RL  = H + (size_t)MM * DFF;       // 4096

    // ---- bf16 weight pool (pre-converted once per launch)
    __bf16* pool = (__bf16*)(RL + MM);
    __bf16* cur  = pool;
    auto take = [&](size_t n) { __bf16* p = cur; cur += n; return p; };

    __bf16* EMBB = take((size_t)VOCAB * DIM);
    __bf16 *WinB[NL], *WABB[NL], *WCB[NL], *WDB[NL], *WoutB[NL],
           *W1B[NL], *W2B[NL], *W3B[NL];
    for (int l = 0; l < NL; ++l) {
        WinB[l]  = take((size_t)DIM * DIM);
        WABB[l]  = take((size_t)DSTATE * (DIM + DSTATE));
        WCB[l]   = take((size_t)DIM * DSTATE);
        WDB[l]   = take((size_t)DIM * DIM);
        WoutB[l] = take((size_t)DIM * DIM);
        W1B[l]   = take((size_t)DFF * DIM);
        W2B[l]   = take((size_t)DFF * DIM);
        W3B[l]   = take((size_t)DIM * DFF);
    }

    float* logits = (float*)d_out;
    float* lossp  = logits + (size_t)MM * VOCAB;

    auto conv = [&](const float* src, __bf16* dst, size_t n) {
        int n4 = (int)(n / 4);
        hipLaunchKernelGGL(conv_kernel, dim3((n4 + 255) / 256), dim3(256), 0, stream,
                           src, dst, n4);
    };
    auto gemm = [&](int mode, const float* A, const __bf16* W, int ldw,
                    const float* bias, const float* extra, float* C, int N, int K) {
        dim3 grid(N / BN, MM / BM);
        switch (mode) {
        case 0: hipLaunchKernelGGL((gemm_kernel<0>), grid, dim3(256), 0, stream,
                                   A, W, ldw, bias, extra, C, MM, N, K); break;
        case 1: hipLaunchKernelGGL((gemm_kernel<1>), grid, dim3(256), 0, stream,
                                   A, W, ldw, bias, extra, C, MM, N, K); break;
        case 2: hipLaunchKernelGGL((gemm_kernel<2>), grid, dim3(256), 0, stream,
                                   A, W, ldw, bias, extra, C, MM, N, K); break;
        default: hipLaunchKernelGGL((gemm_kernel<3>), grid, dim3(256), 0, stream,
                                   A, W, ldw, bias, extra, C, MM, N, K); break;
        }
    };

    // ---- weight pre-conversion pass (fp32 -> bf16, halves GEMM weight traffic)
    conv(emb, EMBB, (size_t)VOCAB * DIM);
    for (int l = 0; l < NL; ++l) {
        const int base = 4 + l * 16;
        conv((const float*)d_in[base + 0],  WinB[l],  (size_t)DIM * DIM);
        conv((const float*)d_in[base + 2],  WABB[l],  (size_t)DSTATE * (DIM + DSTATE));
        conv((const float*)d_in[base + 4],  WCB[l],   (size_t)DIM * DSTATE);
        conv((const float*)d_in[base + 6],  WDB[l],   (size_t)DIM * DIM);
        conv((const float*)d_in[base + 8],  WoutB[l], (size_t)DIM * DIM);
        conv((const float*)d_in[base + 13], W1B[l],   (size_t)DFF * DIM);
        conv((const float*)d_in[base + 14], W2B[l],   (size_t)DFF * DIM);
        conv((const float*)d_in[base + 15], W3B[l],   (size_t)DIM * DFF);
    }

    // x = emb[seq]
    hipLaunchKernelGGL(embed_kernel, dim3(MM), dim3(128), 0, stream, seq, emb, X);

    for (int l = 0; l < NL; ++l) {
        const int base = 4 + l * 16;
        const float* bin  = (const float*)d_in[base + 1];
        const float* bAB  = (const float*)d_in[base + 3];
        const float* bC   = (const float*)d_in[base + 5];
        const float* bD   = (const float*)d_in[base + 7];
        const float* bout = (const float*)d_in[base + 9];
        const float* wln  = (const float*)d_in[base + 10];
        const float* ln1  = (const float*)d_in[base + 11];
        const float* ln2  = (const float*)d_in[base + 12];

        // ---- Mamba block
        hipLaunchKernelGGL(rmsnorm_kernel, dim3(MM), dim3(256), 0, stream, X, ln1, XN);
        gemm(0, XN, WinB[l], DIM, bin, nullptr, U, DIM, DIM);              // u = in_proj
        gemm(0, U, WABB[l] + DSTATE, DIM + DSTATE, bAB, nullptr,
             U2, DSTATE, DIM);                                             // x-part of WAB + bAB
        hipLaunchKernelGGL(scan_kernel, dim3(1), dim3(1024), 0, stream,
                           U2, WABB[l], DIM + DSTATE, wln, LAT);           // recurrence
        gemm(0, LAT, WCB[l], DSTATE, bC, nullptr, T3, DIM, DSTATE);        // ys = lat @ WC^T + bC
        gemm(1, U, WDB[l], DIM, bD, T3, U2, DIM, DIM);                     // ys += u @ WD^T + bD
        gemm(1, U2, WoutB[l], DIM, bout, X, X, DIM, DIM);                  // x += out_proj

        // ---- FFN (SwiGLU)
        hipLaunchKernelGGL(rmsnorm_kernel, dim3(MM), dim3(256), 0, stream, X, ln2, XN);
        gemm(2, XN, W1B[l], DIM, nullptr, nullptr, H, DFF, DIM);           // h = silu(x@w1^T)
        gemm(3, XN, W2B[l], DIM, nullptr, H, H, DFF, DIM);                 // g = h * (x@w2^T)
        gemm(1, H, W3B[l], DFF, nullptr, X, X, DIM, DFF);                  // x += g @ w3^T
    }

    // ---- head: logits + loss
    hipLaunchKernelGGL(rmsnorm_kernel, dim3(MM), dim3(256), 0, stream, X, ln_f, XN);
    gemm(0, XN, EMBB, DIM, nullptr, nullptr, logits, VOCAB, DIM);
    hipLaunchKernelGGL(rowloss_kernel, dim3(MM), dim3(256), 0, stream,
                       logits, targets, RL);
    hipLaunchKernelGGL(loss_reduce_kernel, dim3(1), dim3(256), 0, stream, RL, lossp);
}
